// CrossAttentionNoProj_29454885716533
// MI455X (gfx1250) — matrix operand order repaired and checked by reading, not compiled, not run on hardware
//
#include <hip/hip_runtime.h>

// Problem constants (match reference)
#define BB 2
#define NXQ 512
#define NCC 4096
#define DD 1024
#define HH 16
#define HD 64
#define NK (NXQ + NCC)          // 4608 keys (q appended + context)
#define LN_EPS 1e-5f
#define ATT_SCALE 0.125f        // HD^-0.5

typedef __bf16 bf16_t;
typedef __attribute__((ext_vector_type(16))) __bf16 v16bf;
typedef __attribute__((ext_vector_type(8)))  __bf16 v8bf;
typedef __attribute__((ext_vector_type(8)))  float  v8f;

static __device__ __forceinline__ v8f wmma_bf16(v16bf a, v16bf b, v8f c) {
  // D = A(16x32) * B(32x16) + C, f32 accumulate
  return __builtin_amdgcn_wmma_f32_16x16x32_bf16(false, a, false, b, (short)0, c, false, false);
}

// A operand (16x32 bf16). Layout: lanes 0-15 hold rows 0-15 with k in
// {0..15}; lanes 16-31 hold the same rows with k in {16..31}.
// p points at element [row0][k0] of a row-major matrix, leading dim ld (elems).
static __device__ __forceinline__ v16bf load_a_tile(const bf16_t* p, int ld, int lane) {
  const int row  = lane & 15;
  const int koff = (lane >> 4) * 16;
  v8bf lo = *(const v8bf*)(p + (size_t)row * ld + koff);
  v8bf hi = *(const v8bf*)(p + (size_t)row * ld + koff + 8);
  v16bf a;
#pragma unroll
  for (int i = 0; i < 8; ++i) { a[i] = lo[i]; a[i + 8] = hi[i]; }
  return a;
}

// B operand (32x16 bf16). Layout used here (same k order as load_a_tile): lane holds column n = lane&15, with
// k = (lane>>4)*16 .. +15 contiguous. Source must be stored so that each B
// column n has its 32 k-values contiguous: p + n*ld + k  (32B-aligned load).
static __device__ __forceinline__ v16bf load_b_tile(const bf16_t* p, int ld, int lane) {
  const int col   = lane & 15;
  const int khalf = (lane >> 4) * 16;
  return *(const v16bf*)(p + (size_t)col * ld + khalf);
}

// ---------------------------------------------------------------- LayerNorm
__global__ void ln_kernel(const float* __restrict__ x, const float* __restrict__ w,
                          const float* __restrict__ b, bf16_t* __restrict__ xn) {
  const int row = blockIdx.x;                 // b*NXQ + nx
  const int t   = threadIdx.x;                // 256 threads, 4 elems each
  const float* xr = x + (size_t)row * DD;
  float v[4]; float s = 0.f, s2 = 0.f;
#pragma unroll
  for (int i = 0; i < 4; ++i) { v[i] = xr[t + i * 256]; s += v[i]; s2 += v[i] * v[i]; }
  __shared__ float r1[256], r2[256];
  r1[t] = s; r2[t] = s2; __syncthreads();
  for (int off = 128; off > 0; off >>= 1) {
    if (t < off) { r1[t] += r1[t + off]; r2[t] += r2[t + off]; }
    __syncthreads();
  }
  const float mean = r1[0] * (1.f / DD);
  const float var  = r2[0] * (1.f / DD) - mean * mean;
  const float inv  = rsqrtf(var + LN_EPS);
#pragma unroll
  for (int i = 0; i < 4; ++i) {
    const int idx = t + i * 256;
    xn[(size_t)row * DD + idx] = (bf16_t)((v[i] - mean) * inv * w[idx] + b[idx]);
  }
}

// ------------------------------------------- W (f32 KxN) -> W^T (bf16 NxK)
__global__ void wt_kernel(const float* __restrict__ W, bf16_t* __restrict__ WT) {
  __shared__ float tile[32][33];
  const int x0 = blockIdx.x * 32, y0 = blockIdx.y * 32;
  const int tx = threadIdx.x, ty = threadIdx.y;         // 32 x 8
#pragma unroll
  for (int i = 0; i < 4; ++i)
    tile[ty + i * 8][tx] = W[(size_t)(y0 + ty + i * 8) * DD + x0 + tx];
  __syncthreads();
#pragma unroll
  for (int i = 0; i < 4; ++i)
    WT[(size_t)(x0 + ty + i * 8) * DD + y0 + tx] = (bf16_t)tile[tx][ty + i * 8];
}

// ---- context fan-out: c -> kv0 (f32 out), kv bf16 [bh][key][d], kvT bf16 [bh][d][key]
__global__ void ctx_kernel(const float* __restrict__ c, float* __restrict__ kv0,
                           bf16_t* __restrict__ kv, bf16_t* __restrict__ kvT) {
  const size_t tid = (size_t)blockIdx.x * blockDim.x + threadIdx.x;  // < B*NC*D = 2^23
  const int d = (int)(tid & 63);
  const int h = (int)((tid >> 6) & 15);
  const int n = (int)((tid >> 10) & 4095);
  const int b = (int)(tid >> 22);
  const float v = c[tid];                                  // c[b][n][h][d]
  const int bh = b * HH + h;
  kv0[((size_t)bh * NCC + n) * HD + d]        = v;         // (B,H,NC,HD) output
  kv [((size_t)bh * NK + NXQ + n) * HD + d]   = (bf16_t)v;
  kvT[((size_t)bh * HD + d) * NK + NXQ + n]   = (bf16_t)v;
}

// ------------------------- Q = xn @ Wq, scattered into kv/kvT (first NXQ rows)
__global__ void gemmq_kernel(const bf16_t* __restrict__ A, const bf16_t* __restrict__ BT,
                             bf16_t* __restrict__ kv, bf16_t* __restrict__ kvT) {
  const int lane = threadIdx.x & 31;
  const int wave = (blockIdx.x * blockDim.x + threadIdx.x) >> 5;   // 0..4095
  const int mT = wave >> 6, nT = wave & 63;
  const bf16_t* ap = A  + (size_t)(mT * 16) * DD;
  const bf16_t* bp = BT + (size_t)(nT * 16) * DD;
  v8f acc = {0.f,0.f,0.f,0.f,0.f,0.f,0.f,0.f};
#pragma unroll 4
  for (int k = 0; k < DD; k += 32)
    acc = wmma_bf16(load_a_tile(ap + k, DD, lane), load_b_tile(bp + k, DD, lane), acc);
  const int half = lane >> 4;
  const int n = nT * 16 + (lane & 15);
  const int h = n >> 6, d = n & 63;
#pragma unroll
  for (int r = 0; r < 8; ++r) {
    const int gm = mT * 16 + r + half * 8;      // = b*NXQ + m
    const int b = gm >> 9, m = gm & 511;
    const int bh = b * HH + h;
    const bf16_t qv = (bf16_t)acc[r];
    kv [((size_t)bh * NK + m) * HD + d] = qv;
    kvT[((size_t)bh * HD + d) * NK + m] = qv;
  }
}

// ----------------------------------------------- flash attention, 16 q-rows/wave
#define ATTN_WPB 4
__global__ void attn_kernel(const bf16_t* __restrict__ kv, const bf16_t* __restrict__ kvT,
                            const unsigned char* __restrict__ mask,  // numpy bool, 1B
                            bf16_t* __restrict__ aout) {
  __shared__ __align__(16) bf16_t plds[ATTN_WPB][16 * 32];
  const int lane = threadIdx.x & 31;
  const int w    = threadIdx.x >> 5;
  const int unit = blockIdx.x * ATTN_WPB + w;       // 0..1023
  const int qt = unit & 31;                         // NXQ/16 = 32
  const int bh = unit >> 5;                         // 0..31
  const int b = bh >> 4, h = bh & 15;
  const int q0 = qt * 16;
  const bf16_t* kv_bh  = kv  + (size_t)bh * NK * HD;
  const bf16_t* kvT_bh = kvT + (size_t)bh * HD * NK;
  // Q tile (queries are kv rows q0..q0+15), K split 0..31 / 32..63
  const v16bf qa0 = load_a_tile(kv_bh + (size_t)q0 * HD,      HD, lane);
  const v16bf qa1 = load_a_tile(kv_bh + (size_t)q0 * HD + 32, HD, lane);
  const int half = lane >> 4;
  const int nl   = lane & 15;
  float m_i[8], l_i[8];
  v8f o0 = {0.f,0.f,0.f,0.f,0.f,0.f,0.f,0.f}, o1 = o0, o2 = o0, o3 = o0;
#pragma unroll
  for (int r = 0; r < 8; ++r) { m_i[r] = -3.0e38f; l_i[r] = 0.f; }

  for (int k0 = 0; k0 < NK; k0 += 32) {
    __builtin_prefetch(kv_bh + (size_t)(k0 + 32) * HD, 0, 1);   // next key chunk
    float s[2][8];
    const bool isctx = (k0 >= NXQ);                 // 32-key chunks never straddle
#pragma unroll
    for (int t = 0; t < 2; ++t) {
      const bf16_t* bp = kv_bh + (size_t)(k0 + t * 16) * HD;
      v8f acc = {0.f,0.f,0.f,0.f,0.f,0.f,0.f,0.f};
      acc = wmma_bf16(qa0, load_b_tile(bp,      HD, lane), acc);
      acc = wmma_bf16(qa1, load_b_tile(bp + 32, HD, lane), acc);
      const int key = k0 + t * 16 + nl;
#pragma unroll
      for (int r = 0; r < 8; ++r) {
        float vv = acc[r] * ATT_SCALE;
        if (isctx) {
          const int row = q0 + r + half * 8;
          if (!mask[(size_t)(b * NXQ + row) * NCC + (key - NXQ)]) vv = -3.0e38f;
        }
        s[t][r] = vv;
      }
    }
    // online softmax: row r lives across 16 lanes of this half-wave
    float p0[8], p1[8], alpha[8];
#pragma unroll
    for (int r = 0; r < 8; ++r) {
      float mx = fmaxf(s[0][r], s[1][r]);
#pragma unroll
      for (int sh = 1; sh <= 8; sh <<= 1) mx = fmaxf(mx, __shfl_xor(mx, sh, 32));
      const float mnew = fmaxf(m_i[r], mx);
      alpha[r] = __expf(m_i[r] - mnew);
      p0[r] = __expf(s[0][r] - mnew);
      p1[r] = __expf(s[1][r] - mnew);
      float rs = p0[r] + p1[r];
#pragma unroll
      for (int sh = 1; sh <= 8; sh <<= 1) rs += __shfl_xor(rs, sh, 32);
      l_i[r] = l_i[r] * alpha[r] + rs;
      m_i[r] = mnew;
    }
#pragma unroll
    for (int r = 0; r < 8; ++r) {
      o0[r] *= alpha[r]; o1[r] *= alpha[r]; o2[r] *= alpha[r]; o3[r] *= alpha[r];
    }
    // stage P (16x32, C layout) to per-wave LDS, reload in A layout.
    // Same-wave DS ops are in-order -> no barrier needed.
#pragma unroll
    for (int r = 0; r < 8; ++r) {
      const int row = r + half * 8;
      plds[w][row * 32 + nl]      = (bf16_t)p0[r];
      plds[w][row * 32 + 16 + nl] = (bf16_t)p1[r];
    }
    const int arow = lane & 15, koff = half * 16;
    v8bf plo = *(const v8bf*)&plds[w][arow * 32 + koff];
    v8bf phi = *(const v8bf*)&plds[w][arow * 32 + koff + 8];
    v16bf pa;
#pragma unroll
    for (int i = 0; i < 8; ++i) { pa[i] = plo[i]; pa[i + 8] = phi[i]; }
    // O += P(16x32) * V(32x64): V columns come k-contiguous from kvT
    o0 = wmma_bf16(pa, load_b_tile(kvT_bh + (size_t) 0 * NK + k0, NK, lane), o0);
    o1 = wmma_bf16(pa, load_b_tile(kvT_bh + (size_t)16 * NK + k0, NK, lane), o1);
    o2 = wmma_bf16(pa, load_b_tile(kvT_bh + (size_t)32 * NK + k0, NK, lane), o2);
    o3 = wmma_bf16(pa, load_b_tile(kvT_bh + (size_t)48 * NK + k0, NK, lane), o3);
  }
  // normalize + write a[(b,nx),h*64+d] bf16 for the output GEMM
#pragma unroll
  for (int r = 0; r < 8; ++r) {
    const float inv = 1.f / l_i[r];
    const int row = q0 + r + half * 8;
    bf16_t* orow = aout + (size_t)(b * NXQ + row) * DD + h * HD;
    orow[ 0 + nl] = (bf16_t)(o0[r] * inv);
    orow[16 + nl] = (bf16_t)(o1[r] * inv);
    orow[32 + nl] = (bf16_t)(o2[r] * inv);
    orow[48 + nl] = (bf16_t)(o3[r] * inv);
  }
}

// --------------------------------------------------- o = a @ Wo (f32 output)
__global__ void gemmo_kernel(const bf16_t* __restrict__ A, const bf16_t* __restrict__ BT,
                             float* __restrict__ out) {
  const int lane = threadIdx.x & 31;
  const int wave = (blockIdx.x * blockDim.x + threadIdx.x) >> 5;
  const int mT = wave >> 6, nT = wave & 63;
  const bf16_t* ap = A  + (size_t)(mT * 16) * DD;
  const bf16_t* bp = BT + (size_t)(nT * 16) * DD;
  v8f acc = {0.f,0.f,0.f,0.f,0.f,0.f,0.f,0.f};
#pragma unroll 4
  for (int k = 0; k < DD; k += 32)
    acc = wmma_bf16(load_a_tile(ap + k, DD, lane), load_b_tile(bp + k, DD, lane), acc);
  const int half = lane >> 4;
  const int n = nT * 16 + (lane & 15);
#pragma unroll
  for (int r = 0; r < 8; ++r) {
    const int gm = mT * 16 + r + half * 8;
    out[(size_t)gm * DD + n] = acc[r];
  }
}

extern "C" void kernel_launch(void* const* d_in, const int* in_sizes, int n_in,
                              void* d_out, int out_size, void* d_ws, size_t ws_size,
                              hipStream_t stream) {
  (void)in_sizes; (void)n_in; (void)out_size; (void)ws_size;
  const float* x   = (const float*)d_in[0];
  const float* c   = (const float*)d_in[1];
  const unsigned char* mask = (const unsigned char*)d_in[2];  // bool array (1 byte)
  const float* lnw = (const float*)d_in[3];
  const float* lnb = (const float*)d_in[4];
  const float* Wq  = (const float*)d_in[5];
  const float* Wo  = (const float*)d_in[6];

  float* o_out   = (float*)d_out;                       // (B,NX,D)
  float* kv0_out = o_out + (size_t)BB * NXQ * DD;       // (B,H,NC,HD)

  // workspace carve (all 256B aligned): ~44 MB total
  char*  ws  = (char*)d_ws;
  size_t off = 0;
  auto carve = [&](size_t bytes) {
    char* p = ws + off;
    off += (bytes + 255) & ~(size_t)255;
    return p;
  };
  bf16_t* xn   = (bf16_t*)carve((size_t)BB * NXQ * DD * 2);
  bf16_t* WqT  = (bf16_t*)carve((size_t)DD * DD * 2);
  bf16_t* WoT  = (bf16_t*)carve((size_t)DD * DD * 2);
  bf16_t* kv   = (bf16_t*)carve((size_t)BB * HH * NK * HD * 2);
  bf16_t* kvT  = (bf16_t*)carve((size_t)BB * HH * NK * HD * 2);
  bf16_t* a_bf = (bf16_t*)carve((size_t)BB * NXQ * DD * 2);

  ln_kernel<<<BB * NXQ, 256, 0, stream>>>(x, lnw, lnb, xn);

  dim3 tb(32, 8), tg(DD / 32, DD / 32);
  wt_kernel<<<tg, tb, 0, stream>>>(Wq, WqT);
  wt_kernel<<<tg, tb, 0, stream>>>(Wo, WoT);

  ctx_kernel<<<(BB * NCC * DD) / 256, 256, 0, stream>>>(c, kv0_out, kv, kvT);

  gemmq_kernel<<<1024, 128, 0, stream>>>(xn, WqT, kv, kvT);   // 4096 waves (64x64 tiles)

  attn_kernel<<<(BB * HH * (NXQ / 16)) / ATTN_WPB, 32 * ATTN_WPB, 0, stream>>>(
      kv, kvT, mask, a_bf);

  gemmo_kernel<<<1024, 128, 0, stream>>>(a_bf, WoT, o_out);
}
